// EncoderDecoderRVQ_31602369364119
// MI455X (gfx1250) — compile-verified, matrix operand order checked by reading, not run on hardware
//
#include <hip/hip_runtime.h>
#include <stdint.h>
#include <stddef.h>

// ---------------------------------------------------------------------------
// EncoderDecoderRVQ for MI455X (gfx1250, wave32, WMMA bf16 16x16x32)
//   encoded  = x @ enc_w^T                     (bf16 WMMA GEMM, f32 accum)
//   fused RVQ: residual tile LDS-resident across all 8 layers
//              d2 = ||r||^2 - 2 r.cb^T + ||cb||^2 ; argmin; r -= q
//              quantized = encoded - residual_final  (since sum(q_i) = enc-res)
//   recon    = quantized @ dec_w^T             (bf16 WMMA GEMM, f32 accum)
// Round 4: all-8-layer fusion (residual in LDS, ~5x less HBM traffic),
//          unroll-by-2 ping-pong with literal buffer indices (no v_mov
//          shuffles), native (__bf16) casts for packing.
// ---------------------------------------------------------------------------

typedef __attribute__((ext_vector_type(16))) __bf16        v16bf;
typedef __attribute__((ext_vector_type(8)))  float         v8f;
typedef __attribute__((ext_vector_type(8)))  unsigned int  v8u;

#define RVQ_N   32768      // 16 * 2048 rows
#define RVQ_D   768
#define RVQ_K   128        // codes per layer
#define RVQ_L   8          // layers
#define MT      128        // rows (and cols) per block tile
#define KCH     32         // k-chunk = WMMA K for bf16
#define APAD    8          // +8 ushorts (16B) row padding, keeps uint4 align
#define NTHR    256        // 8 waves of 32
#define NCHUNK  (RVQ_D / KCH)   // 24 (even, required by unroll-by-2)
#define LAU     ((KCH + APAD) / 2)     // staged-tile row stride in uints (20)
#define RTU     ((RVQ_D + APAD) / 2)   // resident-tile row stride in uints (388)
#define INV_ND  (1.0f / ((float)RVQ_N * (float)RVQ_D))

static_assert((NCHUNK & 1) == 0 && NCHUNK >= 2, "pipeline assumes even chunks");

#if defined(__has_builtin)
#  if __has_builtin(__builtin_amdgcn_global_load_async_to_lds_b128) && \
      __has_builtin(__builtin_amdgcn_s_wait_asynccnt)
#    define HAVE_ASYNC_LDS 1
#  endif
#endif
#ifndef HAVE_ASYNC_LDS
#  define HAVE_ASYNC_LDS 0
#endif

// --- bf16 helpers (native cast; LLVM lowers f32->bf16 natively on gfx1250) --
__device__ __forceinline__ unsigned short f32_to_bf16(float f) {
    return __builtin_bit_cast(unsigned short, (__bf16)f);
}
__device__ __forceinline__ unsigned int pack2_bf16(float a, float b) {
    return (unsigned int)f32_to_bf16(a) | ((unsigned int)f32_to_bf16(b) << 16);
}
__device__ __forceinline__ float bf16_lo(unsigned int u) {
    return __builtin_bit_cast(float, u << 16);
}
__device__ __forceinline__ float bf16_hi(unsigned int u) {
    return __builtin_bit_cast(float, u & 0xFFFF0000u);
}

// Build one 16x32 bf16 fragment (A-layout per ISA 7.12.2) from an LDS row view.
// lane 0-15 : VGPR v<4 -> K=2v,2v+1 ; v>=4 -> K=16+2(v-4)..  (uint idx 0-3, 8-11)
// lane 16-31: same +8                                        (uint idx 4-7,12-15)
__device__ __forceinline__ v16bf load_frag(const unsigned int* rowU, int half) {
    v8u u;
    const int kb = half * 4;
#pragma unroll
    for (int v = 0; v < 4; ++v) {
        u[v]     = rowU[kb + v];
        u[v + 4] = rowU[kb + 8 + v];
    }
    return __builtin_bit_cast(v16bf, u);
}

// One k-chunk: 8 WMMAs (16 rows x 128 cols), A frag shared, B frags per tile.
__device__ __forceinline__ void wmma_chunk(v8f (&acc)[8],
                                           const unsigned int* aRowU,
                                           const unsigned int* lBu,
                                           int half, int ln) {
    const v16bf af = load_frag(aRowU, half);
#pragma unroll
    for (int t = 0; t < 8; ++t) {
        const v16bf bf = load_frag(lBu + (size_t)(t * 16 + ln) * LAU, half);
        acc[t] = __builtin_amdgcn_wmma_f32_16x16x32_bf16(
            false, af, false, bf, (short)0, acc[t], false, false);
    }
}

struct ARegs { float4 v[4]; };
__device__ __forceinline__ ARegs loadA(const float* p) {
    ARegs r;
    const float4* s = (const float4*)p;
#pragma unroll
    for (int q = 0; q < 4; ++q) r.v[q] = s[q];
    return r;
}
__device__ __forceinline__ void storeA(unsigned int* dst, const ARegs& r) {
#pragma unroll
    for (int q = 0; q < 4; ++q) {
        dst[q * 2 + 0] = pack2_bf16(r.v[q].x, r.v[q].y);
        dst[q * 2 + 1] = pack2_bf16(r.v[q].z, r.v[q].w);
    }
}

// --- B tile staging: 32 bytes/thread, async global->LDS when available ------
#if HAVE_ASYNC_LDS
typedef __attribute__((__vector_size__(4 * sizeof(int)))) int i32x4;
typedef __attribute__((address_space(1))) i32x4 gi32x4;
typedef __attribute__((address_space(3))) i32x4 li32x4;
__device__ __forceinline__ void stageB(const unsigned short* g, unsigned short* l) {
    // gfx1250: generic-LDS addr low 32 bits == LDS offset (ISA 10.2 aperture map)
    gi32x4* gp = (gi32x4*)(uintptr_t)g;
    li32x4* lp = (li32x4*)(unsigned int)(uintptr_t)l;
    __builtin_amdgcn_global_load_async_to_lds_b128(gp, lp, 0, 0);
    __builtin_amdgcn_global_load_async_to_lds_b128(gp, lp, 16, 0);
}
__device__ __forceinline__ void stageB_wait() { __builtin_amdgcn_s_wait_asynccnt(0); }
#else
__device__ __forceinline__ void stageB(const unsigned short* g, unsigned short* l) {
    const uint4* s = (const uint4*)g;
    uint4* d = (uint4*)l;
    d[0] = s[0]; d[1] = s[1];
}
__device__ __forceinline__ void stageB_wait() {}
#endif

// ---------------------------------------------------------------------------
// Generic C[M,NO] = A_f32[M,DK] * Bw_bf16[NO,DK]^T   (encoder / decoder)
// DK/KCH must be even (768/32 = 24).
// ---------------------------------------------------------------------------
__global__ __launch_bounds__(NTHR)
void gemm_bf16_wmma(const float* __restrict__ A,
                    const unsigned short* __restrict__ Bw,
                    float* __restrict__ C, int NO, int DK) {
    __shared__ unsigned short lA[2][MT][KCH + APAD];
    __shared__ unsigned short lB[2][MT][KCH + APAD];

    const int tid  = threadIdx.x;
    const int lane = tid & 31;
    const int wv   = tid >> 5;        // wave 0..7
    const int half = lane >> 4;
    const int ln   = lane & 15;

    const int rowBase = blockIdx.x * MT;
    const int colBase = blockIdx.y * MT;
    const int ldRow   = tid & 127;    // load-phase row
    const int ldCg    = tid >> 7;     // load-phase 16-col group (0/1)

    const float*          aBase = A  + (size_t)(rowBase + ldRow) * DK + ldCg * 16;
    const unsigned short* bBase = Bw + (size_t)(colBase + ldRow) * DK + ldCg * 16;
    unsigned int*   aD0 = (unsigned int*)&lA[0][ldRow][ldCg * 16];
    unsigned int*   aD1 = (unsigned int*)&lA[1][ldRow][ldCg * 16];
    unsigned short* bD0 = &lB[0][ldRow][ldCg * 16];
    unsigned short* bD1 = &lB[1][ldRow][ldCg * 16];
    const unsigned int* aR0 = (const unsigned int*)&lA[0][wv * 16 + ln][0];
    const unsigned int* aR1 = (const unsigned int*)&lA[1][wv * 16 + ln][0];
    const unsigned int* bU0 = (const unsigned int*)&lB[0][0][0];
    const unsigned int* bU1 = (const unsigned int*)&lB[1][0][0];

    const int nch = DK / KCH;         // even by contract

    {   // prologue: chunk 0 -> buffer 0
        ARegs ar = loadA(aBase);
        stageB(bBase, bD0);
        storeA(aD0, ar);
        stageB_wait();
        __syncthreads();
    }

    v8f acc[8];
#pragma unroll
    for (int t = 0; t < 8; ++t)
#pragma unroll
        for (int r = 0; r < 8; ++r) acc[t][r] = 0.0f;

    for (int i = 0; i < nch; i += 2) {
        // chunk i from buf0; stage chunk i+1 -> buf1 (i+1 < nch always)
        ARegs ar1 = loadA(aBase + (size_t)(i + 1) * KCH);
        stageB(bBase + (size_t)(i + 1) * KCH, bD1);
        wmma_chunk(acc, aR0, bU0, half, ln);
        storeA(aD1, ar1);
        stageB_wait();
        __syncthreads();

        // chunk i+1 from buf1; stage chunk i+2 -> buf0 (if any)
        const bool lastPair = (i + 2 >= nch);
        ARegs ar2;
        if (!lastPair) {
            ar2 = loadA(aBase + (size_t)(i + 2) * KCH);
            stageB(bBase + (size_t)(i + 2) * KCH, bD0);
        }
        wmma_chunk(acc, aR1, bU1, half, ln);
        if (!lastPair) {
            storeA(aD0, ar2);
            stageB_wait();
            __syncthreads();
        }
    }

    // C layout per ISA: VGPR r, lanes 0-15 -> M=r, lanes 16-31 -> M=r+8
#pragma unroll
    for (int t = 0; t < 8; ++t) {
        const int col = colBase + t * 16 + ln;
#pragma unroll
        for (int r = 0; r < 8; ++r) {
            const int row = rowBase + wv * 16 + half * 8 + r;
            C[(size_t)row * NO + col] = acc[t][r];
        }
    }
}

// ---------------------------------------------------------------------------
// Fully fused RVQ: one block owns 128 rows; residual lives in LDS (bf16)
// across all 8 layers. Per layer: distance WMMA + argmin + loss + update.
// Finally quantized = encoded - residual_final.
// ---------------------------------------------------------------------------
__global__ __launch_bounds__(NTHR)
void rvq_fused(const float* __restrict__ encoded,      // [N,D] f32 (encoder out)
               float* __restrict__ quantized,          // [N,D] f32 out
               const unsigned short* __restrict__ cbBf,// [L,K,D] bf16
               const float* __restrict__ cbF,          // [L,K,D] f32
               const float* __restrict__ cnormAll,     // [L,K] ||c||^2
               int* __restrict__ idxOut,               // [N,L]
               float* __restrict__ lossPart)           // [L, gridDim.x]
{
    __shared__ unsigned short rT[MT][RVQ_D + APAD];    // 194 KB resident residual
    __shared__ unsigned short lB[2][RVQ_K][KCH + APAD];
    __shared__ float scores[MT][RVQ_K + 1];            // +1: bank spread
    __shared__ float rrPart[MT][2];
    __shared__ float cn[RVQ_K];
    __shared__ int   bestIdx[MT];
    __shared__ float lossRow[MT];

    const int tid  = threadIdx.x;
    const int lane = tid & 31;
    const int wv   = tid >> 5;
    const int half = lane >> 4;
    const int ln   = lane & 15;

    const int rowBase = blockIdx.x * MT;
    const int ldRow   = tid & 127;
    const int ldCg    = tid >> 7;      // half-row owner (384 floats each)

    unsigned short* bD0 = &lB[0][ldRow][ldCg * 16];
    unsigned short* bD1 = &lB[1][ldRow][ldCg * 16];
    const unsigned int* bU0 = (const unsigned int*)&lB[0][0][0];
    const unsigned int* bU1 = (const unsigned int*)&lB[1][0][0];

    // ---- load residual tile (== encoded) into bf16 LDS; per-row ||r||^2 ----
    {
        const float4* src = (const float4*)(encoded + (size_t)(rowBase + ldRow) * RVQ_D + ldCg * 384);
        unsigned int* dst = (unsigned int*)&rT[ldRow][ldCg * 384];
        float s = 0.0f;
#pragma unroll 4
        for (int i = 0; i < 96; ++i) {
            const float4 f = src[i];
            s += f.x * f.x + f.y * f.y + f.z * f.z + f.w * f.w;
            dst[i * 2 + 0] = pack2_bf16(f.x, f.y);
            dst[i * 2 + 1] = pack2_bf16(f.z, f.w);
        }
        rrPart[ldRow][ldCg] = s;
    }
    __syncthreads();

    float rrRow = (tid < MT) ? (rrPart[tid][0] + rrPart[tid][1]) : 0.0f;

    const unsigned int* aRowBase = (const unsigned int*)&rT[wv * 16 + ln][0];

    for (int layer = 0; layer < RVQ_L; ++layer) {
        if (tid < RVQ_K) cn[tid] = cnormAll[layer * RVQ_K + tid];
        const unsigned short* bBase =
            cbBf + (size_t)layer * RVQ_K * RVQ_D + (size_t)ldRow * RVQ_D + ldCg * 16;

        // prologue: codebook chunk 0 -> buf0 (barrier also covers cn[] + rT)
        stageB(bBase, bD0);
        stageB_wait();
        __syncthreads();

        v8f acc[8];
#pragma unroll
        for (int t = 0; t < 8; ++t)
#pragma unroll
            for (int r = 0; r < 8; ++r) acc[t][r] = 0.0f;

        for (int i = 0; i < NCHUNK; i += 2) {
            stageB(bBase + (size_t)(i + 1) * KCH, bD1);          // i+1 < 24 always
            wmma_chunk(acc, aRowBase + (size_t)i * 16, bU0, half, ln);
            stageB_wait();
            __syncthreads();

            const bool lastPair = (i + 2 >= NCHUNK);
            if (!lastPair) stageB(bBase + (size_t)(i + 2) * KCH, bD0);
            wmma_chunk(acc, aRowBase + (size_t)(i + 1) * 16, bU1, half, ln);
            if (!lastPair) {
                stageB_wait();
                __syncthreads();
            }
        }

        // spill scores: score(m, n) = -2 * (r . c)   (cn added during scan)
#pragma unroll
        for (int t = 0; t < 8; ++t)
#pragma unroll
            for (int r = 0; r < 8; ++r)
                scores[wv * 16 + half * 8 + r][t * 16 + ln] = -2.0f * acc[t][r];
        __syncthreads();

        // per-row argmin over 128 codes (first-min tie-break, like jnp.argmin)
        if (tid < MT) {
            float best = 3.402823466e38f;
            int   bi   = 0;
#pragma unroll 4
            for (int k = 0; k < RVQ_K; ++k) {
                const float s = scores[tid][k] + cn[k];
                if (s < best) { best = s; bi = k; }
            }
            bestIdx[tid] = bi;
            idxOut[(size_t)(rowBase + tid) * RVQ_L + layer] = bi;
            // ||q-r||^2 = ||r||^2 + min(-2 r.c + ||c||^2) ; and that IS the
            // next layer's ||r||^2 (r' = r - q), so update rr incrementally.
            lossRow[tid] = 0.25f * (rrRow + best) * INV_ND;
            rrRow += best;
        }
        __syncthreads();

        if (tid == 0) {            // deterministic serial block reduction
            float s = 0.0f;
            for (int m = 0; m < MT; ++m) s += lossRow[m];
            lossPart[(size_t)layer * gridDim.x + blockIdx.x] = s;
        }

        // update resident residual: r -= cb[idx]   (f32 codebook, bf16 store)
        {
            const int ci = bestIdx[ldRow];
            const float4* qv = (const float4*)(cbF + (size_t)layer * RVQ_K * RVQ_D +
                                               (size_t)ci * RVQ_D + ldCg * 384);
            unsigned int* rp = (unsigned int*)&rT[ldRow][ldCg * 384];
#pragma unroll 4
            for (int i = 0; i < 96; ++i) {
                const float4 q = qv[i];
                const unsigned int u0 = rp[i * 2], u1 = rp[i * 2 + 1];
                rp[i * 2 + 0] = pack2_bf16(bf16_lo(u0) - q.x, bf16_hi(u0) - q.y);
                rp[i * 2 + 1] = pack2_bf16(bf16_lo(u1) - q.z, bf16_hi(u1) - q.w);
            }
        }
        __syncthreads();
    }

    // quantized = encoded - residual_final   (sum of all selected codes)
    {
        const float4* ev = (const float4*)(encoded + (size_t)(rowBase + ldRow) * RVQ_D + ldCg * 384);
        const unsigned int* rp = (const unsigned int*)&rT[ldRow][ldCg * 384];
        float4* qo = (float4*)(quantized + (size_t)(rowBase + ldRow) * RVQ_D + ldCg * 384);
#pragma unroll 4
        for (int i = 0; i < 96; ++i) {
            const float4 e = ev[i];
            const unsigned int u0 = rp[i * 2], u1 = rp[i * 2 + 1];
            float4 o;
            o.x = e.x - bf16_lo(u0);
            o.y = e.y - bf16_hi(u0);
            o.z = e.z - bf16_lo(u1);
            o.w = e.w - bf16_hi(u1);
            qo[i] = o;
        }
    }
}

// ---------------------------------------------------------------------------
// Small helpers
// ---------------------------------------------------------------------------
__global__ void cvt_f32_bf16(const float* __restrict__ s, unsigned short* __restrict__ d, size_t n) {
    const size_t stride = (size_t)gridDim.x * blockDim.x;
    for (size_t i = (size_t)blockIdx.x * blockDim.x + threadIdx.x; i < n; i += stride)
        d[i] = f32_to_bf16(s[i]);
}

__global__ void cnorm_kernel(const float* __restrict__ cb, float* __restrict__ cn) {
    const int t = blockIdx.x * blockDim.x + threadIdx.x;   // layer*128 + code
    if (t < RVQ_L * RVQ_K) {
        const float* p = cb + (size_t)t * RVQ_D;
        float s = 0.0f;
        for (int d = 0; d < RVQ_D; ++d) s += p[d] * p[d];
        cn[t] = s;
    }
}

__global__ void loss_reduce(const float* __restrict__ lossPart, float* __restrict__ lossOut,
                            int nblk) {
    const int l = threadIdx.x;
    if (l < RVQ_L) {
        float s = 0.0f;
        for (int b = 0; b < nblk; ++b) s += lossPart[(size_t)l * nblk + b];
        lossOut[l] = s;
    }
}

// ---------------------------------------------------------------------------
// Launch
// ---------------------------------------------------------------------------
extern "C" void kernel_launch(void* const* d_in, const int* in_sizes, int n_in,
                              void* d_out, int out_size, void* d_ws, size_t ws_size,
                              hipStream_t stream) {
    (void)in_sizes; (void)n_in; (void)out_size; (void)ws_size;

    const float* x    = (const float*)d_in[0];   // [16,2048,768]
    const float* encW = (const float*)d_in[1];   // [768,768]
    const float* decW = (const float*)d_in[2];   // [768,768]
    const float* cb   = (const float*)d_in[3];   // [8,128,768]

    float* out     = (float*)d_out;
    float* recon   = out;                                        // N*D f32
    int*   idxOut  = (int*)(out + (size_t)RVQ_N * RVQ_D);        // N*L i32
    float* lossOut = out + (size_t)RVQ_N * RVQ_D + (size_t)RVQ_N * RVQ_L; // L f32

    const int NBLK = RVQ_N / MT;   // 256

    // workspace carve-out
    float* encoded   = (float*)d_ws;                                   // N*D
    float* quantized = encoded + (size_t)RVQ_N * RVQ_D;                // N*D
    unsigned short* encB = (unsigned short*)(quantized + (size_t)RVQ_N * RVQ_D);
    unsigned short* decB = encB + (size_t)RVQ_D * RVQ_D;
    unsigned short* cbB  = decB + (size_t)RVQ_D * RVQ_D;
    float* cnorm    = (float*)(cbB + (size_t)RVQ_L * RVQ_K * RVQ_D);
    float* lossPart = cnorm + RVQ_L * RVQ_K;                           // L*NBLK

    cvt_f32_bf16<<<1024, NTHR, 0, stream>>>(encW, encB, (size_t)RVQ_D * RVQ_D);
    cvt_f32_bf16<<<1024, NTHR, 0, stream>>>(decW, decB, (size_t)RVQ_D * RVQ_D);
    cvt_f32_bf16<<<1024, NTHR, 0, stream>>>(cb,   cbB,  (size_t)RVQ_L * RVQ_K * RVQ_D);
    cnorm_kernel<<<(RVQ_L * RVQ_K + NTHR - 1) / NTHR, NTHR, 0, stream>>>(cb, cnorm);

    // encoder: encoded = x @ enc_w^T
    dim3 gGemm(NBLK, RVQ_D / MT);
    gemm_bf16_wmma<<<gGemm, NTHR, 0, stream>>>(x, encB, encoded, RVQ_D, RVQ_D);

    // fused RVQ: all 8 layers, residual resident in LDS
    rvq_fused<<<NBLK, NTHR, 0, stream>>>(encoded, quantized, cbB, cb, cnorm,
                                         idxOut, lossPart);

    loss_reduce<<<1, 32, 0, stream>>>(lossPart, lossOut, NBLK);

    // decoder: recon = quantized @ dec_w^T
    gemm_bf16_wmma<<<gGemm, NTHR, 0, stream>>>(quantized, decB, recon, RVQ_D, RVQ_D);
}
